// ParallelAttention_8091718386423
// MI455X (gfx1250) — matrix operand order repaired and checked by reading, not compiled, not run on hardware
//
#include <hip/hip_runtime.h>
#include <math.h>

typedef __attribute__((ext_vector_type(16))) __bf16 v16bf;
typedef __attribute__((ext_vector_type(8)))  float  v8f;
typedef __attribute__((ext_vector_type(4)))  unsigned int v4u;
typedef __attribute__((ext_vector_type(4)))  int v4i;

// ---- problem constants ----
#define BB 2
#define TT 2048
#define DD 2048
#define HH 16
#define HD 128
#define MROWS (BB*TT)        // 4096
#define QKVN  (3*DD)         // 6144

// ---- CDNA5 async global->LDS path (guarded; falls back to plain copies) ----
#if defined(__has_builtin)
#if __has_builtin(__builtin_amdgcn_global_load_async_to_lds_b128) && \
    __has_builtin(__builtin_amdgcn_s_wait_asynccnt)
#define HAVE_ASYNC_LDS 1
#endif
#endif
#ifndef HAVE_ASYNC_LDS
#define HAVE_ASYNC_LDS 0
#endif

__device__ __forceinline__ void copy16_g2l(const __bf16* g, __bf16* l) {
#if HAVE_ASYNC_LDS
  // generic->as1: full 64-bit address; generic->as3: low 32 bits are the LDS offset (ISA 10.2)
  __builtin_amdgcn_global_load_async_to_lds_b128(
      (__attribute__((address_space(1))) v4i*)(unsigned long long)(uintptr_t)g,
      (__attribute__((address_space(3))) v4i*)(unsigned int)(uintptr_t)l,
      0, 0);
#else
  *(v4u*)l = *(const v4u*)g;
#endif
}

__device__ __forceinline__ void async_wait_all() {
#if HAVE_ASYNC_LDS
  __builtin_amdgcn_s_wait_asynccnt(0);
#endif
}

// XOR-butterfly reductions within 16-lane groups via ds_swizzle (no addr setup).
__device__ __forceinline__ float swz_max16(float x) {
  x = fmaxf(x, __int_as_float(__builtin_amdgcn_ds_swizzle(__float_as_int(x), 0x041F)));
  x = fmaxf(x, __int_as_float(__builtin_amdgcn_ds_swizzle(__float_as_int(x), 0x081F)));
  x = fmaxf(x, __int_as_float(__builtin_amdgcn_ds_swizzle(__float_as_int(x), 0x101F)));
  x = fmaxf(x, __int_as_float(__builtin_amdgcn_ds_swizzle(__float_as_int(x), 0x201F)));
  return x;
}
__device__ __forceinline__ float swz_sum16(float x) {
  x += __int_as_float(__builtin_amdgcn_ds_swizzle(__float_as_int(x), 0x041F));
  x += __int_as_float(__builtin_amdgcn_ds_swizzle(__float_as_int(x), 0x081F));
  x += __int_as_float(__builtin_amdgcn_ds_swizzle(__float_as_int(x), 0x101F));
  x += __int_as_float(__builtin_amdgcn_ds_swizzle(__float_as_int(x), 0x201F));
  return x;
}

// Load a 16-element bf16 WMMA fragment from two 16-byte LDS chunks.
__device__ __forceinline__ v16bf load_frag2(const __bf16* p0, const __bf16* p1) {
  union { v16bf v; v4u u[2]; } t;
  t.u[0] = *(const v4u*)p0;
  t.u[1] = *(const v4u*)p1;
  return t.v;
}

// ---------------- f32 -> bf16 elementwise (for x) ----------------
__global__ __launch_bounds__(256) void f32_to_bf16_kernel(const float* __restrict__ in,
                                                          __bf16* __restrict__ out, int n4) {
  int i = blockIdx.x * blockDim.x + threadIdx.x;
  if (i >= n4) return;
  const float4 f = *(const float4*)(in + (size_t)i * 4);
  union { __bf16 e[4]; uint2 u; } t;
  t.e[0] = (__bf16)f.x; t.e[1] = (__bf16)f.y; t.e[2] = (__bf16)f.z; t.e[3] = (__bf16)f.w;
  *(uint2*)(out + (size_t)i * 4) = t.u;
}

// ---------------- f32 [rows][cols] -> bf16 [cols][rows] tiled transpose ----------------
__global__ __launch_bounds__(256) void transpose_to_bf16_kernel(const float* __restrict__ in,
                                                                __bf16* __restrict__ out,
                                                                int rows, int cols) {
  __shared__ __bf16 tile[32][33];
  const int c0 = blockIdx.x * 32, r0 = blockIdx.y * 32;
  const int tx = threadIdx.x & 31, ty = threadIdx.x >> 5;   // 32 x 8
  #pragma unroll
  for (int i = 0; i < 4; ++i) {
    int r = ty + i * 8;
    tile[r][tx] = (__bf16)in[(size_t)(r0 + r) * cols + c0 + tx];
  }
  __syncthreads();
  #pragma unroll
  for (int i = 0; i < 4; ++i) {
    int c = ty + i * 8;
    out[(size_t)(c0 + c) * rows + r0 + tx] = tile[tx][c];
  }
}

// ---------------- bf16 GEMM: C[M,N] = A[M,K] @ Bt[N,K]^T ----------------
// Bt is PRE-TRANSPOSED ([N][K] row-major) => both LDS tile fills are straight copies.
// 256x128 block tile, BK=32, double-buffered async LDS fills; 8 waves in 4(m) x 2(n);
// each wave 64x64 = 16 WMMA tiles; 1 barrier per K-step.
template <typename OutT>
__global__ __launch_bounds__(256) void gemm_bf16_kernel(const __bf16* __restrict__ A,
                                                        const __bf16* __restrict__ Bt,
                                                        OutT* __restrict__ C,
                                                        int M, int N, int K) {
  __shared__ __bf16 Alds[2][256 * 40];   // [buf][row][k], padded stride 40
  __shared__ __bf16 Blds[2][128 * 40];   // [buf][n][k],  padded stride 40
  const int tid  = threadIdx.x;
  const int lane = tid & 31;
  const int wid  = tid >> 5;
  const int l16  = lane & 15;
  const int hi   = lane >> 4;            // 0 or 1
  const int wm   = (wid >> 1) * 64;      // wave row base within tile
  const int wn   = (wid & 1) * 64;       // wave col base within tile
  const int bm   = blockIdx.y * 256;
  const int bn   = blockIdx.x * 128;

  auto load_tiles = [&](int buf, int k0) {
    // A tile: 256 rows x 32 cols = 1024 x 16B vectors (4 per thread)
    #pragma unroll
    for (int i = 0; i < 4; ++i) {
      int vec = tid + i * 256;
      int r = vec >> 2;
      int c = (vec & 3) << 3;
      copy16_g2l(&A[(size_t)(bm + r) * K + k0 + c], &Alds[buf][r * 40 + c]);
    }
    // B tile (already [n][k]): 128 rows x 32 cols = 512 x 16B vectors (2 per thread)
    #pragma unroll
    for (int i = 0; i < 2; ++i) {
      int vec = tid + i * 256;
      int n = vec >> 2;
      int c = (vec & 3) << 3;
      copy16_g2l(&Bt[(size_t)(bn + n) * K + k0 + c], &Blds[buf][n * 40 + c]);
    }
  };

  v8f acc[4][4] = {};
  const int nk = K >> 5;
  load_tiles(0, 0);

  for (int kt = 0; kt < nk; ++kt) {
    async_wait_all();
    __syncthreads();
    if (kt + 1 < nk) load_tiles((kt + 1) & 1, (kt + 1) << 5);

    const __bf16* Al = Alds[kt & 1];
    const __bf16* Bl = Blds[kt & 1];
    v16bf bfr[4];
    #pragma unroll
    for (int ns = 0; ns < 4; ++ns) {
      int n = wn + ns * 16 + l16;
      int kk = hi ? 8 : 0;               // B frag: {0-7,16-23} / {8-15,24-31}
      bfr[ns] = load_frag2(&Bl[n * 40 + kk], &Bl[n * 40 + kk + 16]);
    }
    #pragma unroll
    for (int ms = 0; ms < 4; ++ms) {
      int r  = wm + ms * 16 + l16;
      int kk = hi ? 8 : 0;               // A frag: {0-7,16-23} / {8-15,24-31}
      v16bf af = load_frag2(&Al[r * 40 + kk], &Al[r * 40 + kk + 16]);
      #pragma unroll
      for (int ns = 0; ns < 4; ++ns)
        acc[ms][ns] = __builtin_amdgcn_wmma_f32_16x16x32_bf16(
            false, af, false, bfr[ns], (short)0, acc[ms][ns], false, false);
    }
  }

  // Epilogue: D layout -> lane owns column l16; VGPR r holds row r (+8 for hi lanes)
  #pragma unroll
  for (int ms = 0; ms < 4; ++ms) {
    #pragma unroll
    for (int ns = 0; ns < 4; ++ns) {
      int gn    = bn + wn + ns * 16 + l16;
      int rbase = bm + wm + ms * 16 + hi * 8;
      #pragma unroll
      for (int r = 0; r < 8; ++r)
        C[(size_t)(rbase + r) * N + gn] = (OutT)acc[ms][ns][r];
    }
  }
}

// ---------------- flash attention ----------------
// grid: (T/128, B*H). 256 threads = 8 waves; wave w owns queries qb*128+w*16 .. +15.
// Double-buffered K/V tiles (async K fills), 1 barrier per key tile.
__global__ __launch_bounds__(256) void flash_attn_kernel(const __bf16* __restrict__ qkv,
                                                         __bf16* __restrict__ y) {
  const int qb  = blockIdx.x;
  const int bh  = blockIdx.y;
  const int b   = bh >> 4;
  const int h   = bh & 15;
  const int tid = threadIdx.x;
  const int lane = tid & 31;
  const int wid  = tid >> 5;
  const int l16  = lane & 15;
  const int hi   = lane >> 4;

  __shared__ __bf16 Klds[2][32 * 136];   // [buf][key][hd]: == B-frag [N][K] for Q*K^T
  __shared__ __bf16 Vlds[2][128 * 40];   // [buf][hd][key]: == B-frag [N][K] for P*V
  __shared__ __bf16 Plds[8][16 * 40];    // per-wave P staging (D-layout -> A-layout)

  const size_t rowbase = (size_t)b * TT;
  const int qstart = qb * 128 + wid * 16;
  const int qend   = qstart + 15;
  const int qcol = h * HD, kcol = DD + h * HD, vcol = 2 * DD + h * HD;
  // exp(x*scale) == exp2(x * scale * log2(e)); track max on UNSCALED scores
  const float cexp = 0.088388347648318447f * 1.4426950408889634f;

  auto load_kv = [&](int buf, int kb) {
    // K tile: straight [key][hd] copy (async path if available)
    #pragma unroll
    for (int i = 0; i < 2; ++i) {
      int vec = tid + i * 256;
      int kk = vec >> 4;
      int c  = (vec & 15) << 3;
      copy16_g2l(&qkv[(rowbase + kb + kk) * QKVN + kcol + c], &Klds[buf][kk * 136 + c]);
    }
    // V tile: transpose into [hd][key]
    #pragma unroll
    for (int i = 0; i < 2; ++i) {
      int vec = tid + i * 256;
      int kk = vec >> 4;
      int c  = (vec & 15) << 3;
      union { v4u u; __bf16 e[8]; } t;
      t.u = *(const v4u*)&qkv[(rowbase + kb + kk) * QKVN + vcol + c];
      #pragma unroll
      for (int j = 0; j < 8; ++j) Vlds[buf][(c + j) * 40 + kk] = t.e[j];
    }
  };

  // Preload Q fragments (16 queries x 128 hd = 4 A-frags of 16x32) straight from global
  v16bf qf[4];
  {
    const __bf16* qp = qkv + (rowbase + qstart + l16) * QKVN + qcol;
    const int kofs = hi ? 8 : 0;
    #pragma unroll
    for (int c = 0; c < 4; ++c)
      qf[c] = load_frag2(qp + c * 32 + kofs, qp + c * 32 + kofs + 16);
  }

  float mrow[8], lrow[8];
  #pragma unroll
  for (int r = 0; r < 8; ++r) { mrow[r] = -1e30f; lrow[r] = 0.0f; }
  v8f o[8] = {};

  const int ntiles = (qb + 1) * 4;       // 32-key tiles covering causal range
  load_kv(0, 0);
  for (int kt = 0; kt < ntiles; ++kt) {
    const int kb = kt * 32;
    async_wait_all();
    __syncthreads();
    if (kt + 1 < ntiles) load_kv((kt + 1) & 1, kb + 32);
    if (kb > qend) continue;             // fully masked for this wave

    const __bf16* Kl = Klds[kt & 1];
    const __bf16* Vl = Vlds[kt & 1];

    // S = Q * K^T  (16 q x 32 k: two 16x16 accum tiles)
    v8f s[2] = {};
    #pragma unroll
    for (int ks = 0; ks < 2; ++ks) {
      const int n  = ks * 16 + l16;
      const int kk = hi ? 8 : 0;
      #pragma unroll
      for (int c = 0; c < 4; ++c) {
        v16bf bf = load_frag2(&Kl[n * 136 + c * 32 + kk],
                              &Kl[n * 136 + c * 32 + kk + 16]);
        s[ks] = __builtin_amdgcn_wmma_f32_16x16x32_bf16(
            false, qf[c], false, bf, (short)0, s[ks], false, false);
      }
    }
    // causal mask only when this tile can touch the diagonal
    if (kb + 31 > qstart) {
      #pragma unroll
      for (int ks = 0; ks < 2; ++ks) {
        int kg = kb + ks * 16 + l16;
        #pragma unroll
        for (int r = 0; r < 8; ++r) {
          int qg = qstart + r + hi * 8;
          s[ks][r] = (kg <= qg) ? s[ks][r] : -1e30f;
        }
      }
    }
    // online softmax (row stats live in one VGPR index across each 16-lane group)
    float alpha[8], mnew[8];
    #pragma unroll
    for (int r = 0; r < 8; ++r)
      mnew[r] = fmaxf(mrow[r], swz_max16(fmaxf(s[0][r], s[1][r])));
    #pragma unroll
    for (int ks = 0; ks < 2; ++ks)
      #pragma unroll
      for (int r = 0; r < 8; ++r)
        s[ks][r] = exp2f((s[ks][r] - mnew[r]) * cexp);
    #pragma unroll
    for (int r = 0; r < 8; ++r) {
      float sm = swz_sum16(s[0][r] + s[1][r]);
      alpha[r] = exp2f((mrow[r] - mnew[r]) * cexp);
      lrow[r]  = lrow[r] * alpha[r] + sm;
      mrow[r]  = mnew[r];
    }
    #pragma unroll
    for (int j = 0; j < 8; ++j)
      #pragma unroll
      for (int r = 0; r < 8; ++r)
        o[j][r] *= alpha[r];

    // P: D-layout f32 -> bf16 A-layout via per-wave LDS staging
    __bf16* pl = Plds[wid];
    #pragma unroll
    for (int ks = 0; ks < 2; ++ks)
      #pragma unroll
      for (int r = 0; r < 8; ++r)
        pl[(r + hi * 8) * 40 + ks * 16 + l16] = (__bf16)s[ks][r];
    const int kofs = hi ? 8 : 0;
    v16bf pf = load_frag2(&pl[l16 * 40 + kofs], &pl[l16 * 40 + kofs + 16]);

    // O += P * V
    #pragma unroll
    for (int j = 0; j < 8; ++j) {
      int n  = j * 16 + l16;
      int kk = hi ? 8 : 0;
      v16bf vf = load_frag2(&Vl[n * 40 + kk], &Vl[n * 40 + kk + 16]);
      o[j] = __builtin_amdgcn_wmma_f32_16x16x32_bf16(
          false, pf, false, vf, (short)0, o[j], false, false);
    }
  }

  // normalize and store y (bf16) at [row, h*128 + hd]
  #pragma unroll
  for (int j = 0; j < 8; ++j) {
    #pragma unroll
    for (int r = 0; r < 8; ++r) {
      int qg = qstart + r + hi * 8;
      float val = o[j][r] / lrow[r];
      y[(rowbase + qg) * DD + h * HD + j * 16 + l16] = (__bf16)val;
    }
  }
}

extern "C" void kernel_launch(void* const* d_in, const int* in_sizes, int n_in,
                              void* d_out, int out_size, void* d_ws, size_t ws_size,
                              hipStream_t stream) {
  const float* x      = (const float*)d_in[0];
  const float* w_attn = (const float*)d_in[1];
  const float* w_proj = (const float*)d_in[2];
  float* out = (float*)d_out;

  __bf16* x16  = (__bf16*)d_ws;                         // [4096][2048]
  __bf16* waT  = x16  + (size_t)MROWS * DD;             // [6144][2048]  (w_attn^T)
  __bf16* wpT  = waT  + (size_t)DD * QKVN;              // [2048][2048]  (w_proj^T)
  __bf16* qkv  = wpT  + (size_t)DD * DD;                // [4096][6144]
  __bf16* y16  = qkv  + (size_t)MROWS * QKVN;           // [4096][2048]

  { int n4 = (MROWS * DD) / 4;
    f32_to_bf16_kernel<<<(n4 + 255) / 256, 256, 0, stream>>>(x, x16, n4); }
  // weights: transpose-convert once so GEMM B tiles are straight [N][K] copies
  transpose_to_bf16_kernel<<<dim3(QKVN / 32, DD / 32), 256, 0, stream>>>(w_attn, waT, DD, QKVN);
  transpose_to_bf16_kernel<<<dim3(DD / 32, DD / 32), 256, 0, stream>>>(w_proj, wpT, DD, DD);

  // QKV GEMM: [4096 x 2048] @ [2048 x 6144] -> bf16
  gemm_bf16_kernel<__bf16><<<dim3(QKVN / 128, MROWS / 256), 256, 0, stream>>>(
      x16, waT, qkv, MROWS, QKVN, DD);

  // Attention
  flash_attn_kernel<<<dim3(TT / 128, BB * HH), 256, 0, stream>>>(qkv, y16);

  // Projection: [4096 x 2048] @ [2048 x 2048] -> fp32 out
  gemm_bf16_kernel<float><<<dim3(DD / 128, MROWS / 256), 256, 0, stream>>>(
      y16, wpT, out, MROWS, DD, DD);
}
